// ConvNextBlockLayer_51436528337498
// MI455X (gfx1250) — matrix operand order repaired and checked by reading, not compiled, not run on hardware
//
#include <hip/hip_runtime.h>
#include <hip/hip_bf16.h>
#include <math.h>

typedef __attribute__((ext_vector_type(16))) __bf16 v16bf;
typedef __attribute__((ext_vector_type(8)))  float  v8f;

#define Bsz 16
#define Cch 96
#define Hh  112
#define Ww  112
#define HID 384
#define NT1 24   // 384/16 output tiles GEMM1
#define NT2 6    // 96/16 output tiles GEMM2
#define KK2 12   // 384/32 k-chunks GEMM2
#define MT  7    // 112/16 row tiles
#define NTHREADS 224   // 7 waves: one 16-row M-tile per wave, no idle wave

// LDS layout (dynamic). Strides padded so 16 consecutive rows map to 16
// distinct banks (stride/4 dwords: 49,52 -> odd*4 mod 64 permutes 16 lanes).
#define CONV_STRIDE 97    // f32, dwconv scratch
#define YLN_STRIDE  104   // bf16, LN output (A source, GEMM1)
#define H_STRIDE    392   // bf16, GELU output (A source, GEMM2)
#define W1_STRIDE   104   // bf16, staged W1^T [384][96+pad]
#define W2_STRIDE   392   // bf16, staged W2^T [96][384+pad]

#define OFF_H    0
#define OFF_YLN  (Ww * H_STRIDE * 2)              // 87808
#define OFF_W1   (OFF_YLN + Ww * YLN_STRIDE * 2)  // 111104
#define OFF_W2   (OFF_W1 + HID * W1_STRIDE * 2)   // 190976
#define OFF_B1   (OFF_W2 + Cch * W2_STRIDE * 2)   // 266240  (384 f32)
#define OFF_B2   (OFF_B1 + HID * 4)               // 267776  (96 f32)
#define OFF_SCL  (OFF_B2 + Cch * 4)               // 268160  (96 f32)
#define SMEM_BYTES (OFF_SCL + Cch * 4)            // 268544

__global__ void cvt_weights_bf16(const float* __restrict__ w1,
                                 const float* __restrict__ w2,
                                 __bf16* __restrict__ w1t,   // [384][96]  dense, n-major
                                 __bf16* __restrict__ w2t) { // [96][384]  dense, c-major
    int i = blockIdx.x * blockDim.x + threadIdx.x;
    if (i < Cch * HID) {
        int c = i / HID, n = i - c * HID;        // w1 is [C][HID]
        w1t[n * Cch + c] = (__bf16)w1[i];
        int k = i / Cch, cc = i - k * Cch;       // w2 is [HID][C]
        w2t[cc * HID + k] = (__bf16)w2[i];
    }
}

// Branch-free exact-GELU: erf via Abramowitz-Stegun 7.1.26 (|err|<1.5e-7),
// using native v_rcp_f32 / v_exp_f32 — no divergence inside the WMMA loop.
__device__ __forceinline__ float gelu_exact(float x) {
    float u = 0.70710678118654752f * x;
    float z = __builtin_fabsf(u);
    float t = __builtin_amdgcn_rcpf(__builtin_fmaf(0.3275911f, z, 1.0f));
    float p = __builtin_fmaf(t, 1.061405429f, -1.453152027f);
    p = __builtin_fmaf(t, p, 1.421413741f);
    p = __builtin_fmaf(t, p, -0.284496736f);
    p = __builtin_fmaf(t, p, 0.254829592f);
    p = p * t;
    float e = __builtin_amdgcn_exp2f(-z * z * 1.4426950408889634f);
    float erfz = __builtin_fmaf(-p, e, 1.0f);
    float erfu = __builtin_copysignf(erfz, u);
    return 0.5f * x * (1.0f + erfu);
}

// gfx1250 async global->LDS copy, 16B per lane, tracked by ASYNCcnt.
__device__ __forceinline__ void async_copy16(unsigned int lds_addr, unsigned long long gaddr) {
    asm volatile("global_load_async_to_lds_b128 %0, %1, off"
                 :: "v"(lds_addr), "v"(gaddr) : "memory");
}

__global__ void __launch_bounds__(NTHREADS)
convnext_block_fused(const float* __restrict__ x,
                     const float* __restrict__ dww,   // (7,7,1,C)
                     const float* __restrict__ dwb,
                     const float* __restrict__ gam,
                     const float* __restrict__ bet,
                     const __bf16* __restrict__ w1t,
                     const float* __restrict__ pb1,
                     const __bf16* __restrict__ w2t,
                     const float* __restrict__ pb2,
                     const float* __restrict__ scl,
                     float* __restrict__ out)
{
    extern __shared__ __align__(16) unsigned char smem[];
    __bf16* s_h    = (__bf16*)(smem + OFF_H);    // [112][392]
    float*  s_conv = (float*)(smem + OFF_H);     // [112][97] (aliased, dead after LN)
    __bf16* s_yln  = (__bf16*)(smem + OFF_YLN);  // [112][104]
    const __bf16* sW1  = (const __bf16*)(smem + OFF_W1);  // [384][104]
    const __bf16* sW2  = (const __bf16*)(smem + OFF_W2);  // [96][392]
    const float*  sB1  = (const float*)(smem + OFF_B1);   // [384]
    const float*  sB2  = (const float*)(smem + OFF_B2);   // [96]
    const float*  sSCL = (const float*)(smem + OFF_SCL);  // [96]

    const int tid  = threadIdx.x;
    const int lane = tid & 31;
    const int wv   = tid >> 5;      // 7 waves, wave w owns M-tile w
    const int hi   = lane >> 4;     // lane half (0/1)
    const int ln   = lane & 15;

    const int bimg = blockIdx.x / Hh;
    const int h0   = blockIdx.x - bimg * Hh;
    const float* xb = x + (size_t)bimg * (Cch * Hh * Ww);

    // ------- Phase 0: async-stage weights + biases into LDS (ASYNCcnt path) -------
    {
        unsigned int l1 = (unsigned int)(uintptr_t)(smem + OFF_W1);
        unsigned int l2 = (unsigned int)(uintptr_t)(smem + OFF_W2);
        // W1t dense [384][96]: 12 x 16B chunks per row -> padded stride 208B
        for (int i = tid; i < HID * 12; i += NTHREADS) {
            int n = i / 12, ch = i - n * 12;
            async_copy16(l1 + n * (W1_STRIDE * 2) + ch * 16,
                         (unsigned long long)(uintptr_t)((const char*)w1t + i * 16));
        }
        // W2t dense [96][384]: 48 x 16B chunks per row -> padded stride 784B
        for (int i = tid; i < Cch * 48; i += NTHREADS) {
            int c = i / 48, ch = i - c * 48;
            async_copy16(l2 + c * (W2_STRIDE * 2) + ch * 16,
                         (unsigned long long)(uintptr_t)((const char*)w2t + i * 16));
        }
        // biases + layerscale (so GEMM loops never touch global for scalars)
        unsigned int lb1 = (unsigned int)(uintptr_t)(smem + OFF_B1);
        unsigned int lb2 = (unsigned int)(uintptr_t)(smem + OFF_B2);
        unsigned int lsc = (unsigned int)(uintptr_t)(smem + OFF_SCL);
        if (tid < HID / 4)
            async_copy16(lb1 + tid * 16, (unsigned long long)(uintptr_t)((const char*)pb1 + tid * 16));
        if (tid < Cch / 4) {
            async_copy16(lb2 + tid * 16, (unsigned long long)(uintptr_t)((const char*)pb2 + tid * 16));
            async_copy16(lsc + tid * 16, (unsigned long long)(uintptr_t)((const char*)scl + tid * 16));
        }
    }

    // ---------------- Phase 1a: depthwise 7x7 conv (SAME, zero pad) -------------
    for (int i = tid; i < Cch * Ww; i += NTHREADS) {   // 10752 = 48 * 224, no tail
        int c = i / Ww;
        int w = i - c * Ww;
        const float* xc = xb + (size_t)c * (Hh * Ww);
        float acc = dwb[c];
        #pragma unroll
        for (int dh = 0; dh < 7; ++dh) {
            int hh = h0 - 3 + dh;
            if ((unsigned)hh < (unsigned)Hh) {
                const float* xr = xc + hh * Ww;
                const float* wr = dww + (dh * 7) * Cch + c;
                #pragma unroll
                for (int dw = 0; dw < 7; ++dw) {
                    int ww = w - 3 + dw;
                    if ((unsigned)ww < (unsigned)Ww)
                        acc = fmaf(xr[ww], wr[dw * Cch], acc);
                }
            }
        }
        s_conv[w * CONV_STRIDE + c] = acc;
    }
    __syncthreads();

    // ---------------- Phase 1b: LayerNorm over channels, write bf16 --------------
    if (tid < Ww) {
        const float* yp = s_conv + tid * CONV_STRIDE;
        float s = 0.f, ss = 0.f;
        #pragma unroll 8
        for (int c = 0; c < Cch; ++c) { float v = yp[c]; s += v; ss += v * v; }
        float mu  = s * (1.0f / Cch);
        float var = ss * (1.0f / Cch) - mu * mu;
        float rs  = __builtin_amdgcn_rsqf(var + 1e-6f);
        __bf16* op = s_yln + tid * YLN_STRIDE;
        #pragma unroll 8
        for (int c = 0; c < Cch; ++c)
            op[c] = (__bf16)((yp[c] - mu) * rs * gam[c] + bet[c]);
    }
    // all async stages must have landed before GEMM1 reads sW1/sB1
    asm volatile("s_wait_asynccnt 0" ::: "memory");
    __syncthreads();

    // ---------------- GEMM1: [16x96] @ [96x384] + b1 -> GELU -> s_h (bf16) -------
    const int mt = wv;   // one M-tile per wave, all 7 waves busy
    {
        // A fragments (K = 0..95 in 3 chunks of 32), contiguous K: k = 16*hi + e
        v16bf a0, a1, a2;
        {
            const __bf16* row = s_yln + (mt * 16 + ln) * YLN_STRIDE;
            #pragma unroll
            for (int e = 0; e < 8; ++e) {
                a0[e]     = row[hi * 16 + e];
                a0[e + 8] = row[hi * 16 + 8 + e];
                a1[e]     = row[32 + hi * 16 + e];
                a1[e + 8] = row[32 + hi * 16 + 8 + e];
                a2[e]     = row[64 + hi * 16 + e];
                a2[e + 8] = row[64 + hi * 16 + 8 + e];
            }
        }
        for (int nt = 0; nt < NT1; ++nt) {
            int n = nt * 16 + ln;
            const __bf16* bp = sW1 + n * W1_STRIDE + hi * 16;  // column n, 16 contiguous K
            v16bf b0, bb1, bb2;
            #pragma unroll
            for (int e = 0; e < 16; ++e) {
                b0[e]  = bp[e];
                bb1[e] = bp[32 + e];
                bb2[e] = bp[64 + e];
            }
            float bias = sB1[n];          // ds_load, shares dscnt with B-frag loads
            v8f acc;
            #pragma unroll
            for (int v = 0; v < 8; ++v) acc[v] = bias;
            acc = __builtin_amdgcn_wmma_f32_16x16x32_bf16(false, a0, false, b0,  (short)0, acc, false, false);
            acc = __builtin_amdgcn_wmma_f32_16x16x32_bf16(false, a1, false, bb1, (short)0, acc, false, false);
            acc = __builtin_amdgcn_wmma_f32_16x16x32_bf16(false, a2, false, bb2, (short)0, acc, false, false);
            // exact-GELU epilogue (branch-free); D layout: m = v + 8*hi, n = ln
            __bf16* hp = s_h + (mt * 16 + 8 * hi) * H_STRIDE + n;
            #pragma unroll
            for (int v = 0; v < 8; ++v)
                hp[v * H_STRIDE] = (__bf16)gelu_exact(acc[v]);
        }
    }
    __syncthreads();

    // ---------------- GEMM2: [16x384] @ [384x96] + b2 -> scale*y + x -------------
    {
        v16bf A[KK2];
        {
            const __bf16* row = s_h + (mt * 16 + ln) * H_STRIDE;
            #pragma unroll
            for (int kk = 0; kk < KK2; ++kk) {
                #pragma unroll
                for (int e = 0; e < 8; ++e) {
                    A[kk][e]     = row[kk * 32 + hi * 16 + e];
                    A[kk][e + 8] = row[kk * 32 + hi * 16 + 8 + e];
                }
            }
        }
        for (int nt = 0; nt < NT2; ++nt) {
            int c = nt * 16 + ln;
            float bias = sB2[c];
            v8f acc;
            #pragma unroll
            for (int v = 0; v < 8; ++v) acc[v] = bias;
            const __bf16* bp = sW2 + c * W2_STRIDE + hi * 16;
            #pragma unroll
            for (int kk = 0; kk < KK2; ++kk) {
                v16bf bf;
                #pragma unroll
                for (int e = 0; e < 16; ++e) bf[e] = bp[kk * 32 + e];
                acc = __builtin_amdgcn_wmma_f32_16x16x32_bf16(false, A[kk], false, bf, (short)0, acc, false, false);
            }
            // epilogue: out[b,c,h0, mt*16 + m] = scale[c]*y + x ; m = v + 8*hi
            float sc = sSCL[c];
            size_t base = (((size_t)(bimg * Cch + c) * Hh + h0) * Ww) + mt * 16 + 8 * hi;
            const float4* xr = (const float4*)(x + base);
            float4 x0 = xr[0], x1 = xr[1];
            float4 o0, o1;
            o0.x = sc * acc[0] + x0.x;  o0.y = sc * acc[1] + x0.y;
            o0.z = sc * acc[2] + x0.z;  o0.w = sc * acc[3] + x0.w;
            o1.x = sc * acc[4] + x1.x;  o1.y = sc * acc[5] + x1.y;
            o1.z = sc * acc[6] + x1.z;  o1.w = sc * acc[7] + x1.w;
            float4* op = (float4*)(out + base);
            op[0] = o0;
            op[1] = o1;
        }
    }
}

extern "C" void kernel_launch(void* const* d_in, const int* in_sizes, int n_in,
                              void* d_out, int out_size, void* d_ws, size_t ws_size,
                              hipStream_t stream) {
    const float* x    = (const float*)d_in[0];
    const float* dw_w = (const float*)d_in[1];
    const float* dw_b = (const float*)d_in[2];
    const float* gam  = (const float*)d_in[3];
    const float* bet  = (const float*)d_in[4];
    const float* w1   = (const float*)d_in[5];
    const float* b1   = (const float*)d_in[6];
    const float* w2   = (const float*)d_in[7];
    const float* b2   = (const float*)d_in[8];
    const float* scl  = (const float*)d_in[9];
    float* out = (float*)d_out;

    __bf16* w1t = (__bf16*)d_ws;                 // 384*96 bf16, dense
    __bf16* w2t = w1t + (size_t)Cch * HID;       // 96*384 bf16, dense

    {
        int n = Cch * HID;
        cvt_weights_bf16<<<(n + 255) / 256, 256, 0, stream>>>(w1, w2, w1t, w2t);
    }
    convnext_block_fused<<<Bsz * Hh, NTHREADS, SMEM_BYTES, stream>>>(
        x, dw_w, dw_b, gam, bet, w1t, b1, w2t, b2, scl, out);
}
